// VAE_20959440404731
// MI455X (gfx1250) — matrix operand order repaired and checked by reading, not compiled, not run on hardware
//
#include <hip/hip_runtime.h>
#include <hip/hip_bf16.h>
#include <math.h>

// ---------------- model constants ----------------
#define DMODEL 512
#define SEQL   60
#define VOCAB  2500
#define ZDIM   1024
#define NHEAD  8
#define DHEAD  64
#define FFDIM  2048
#define NLAYER 6
#define BATCH  256
#define NTOK   (BATCH * SEQL)   // 15360 tokens

// ---------------- WMMA types (CDNA5 / gfx1250, wave32) ----------------
typedef __bf16 bf16_t;
typedef bf16_t v16bf __attribute__((ext_vector_type(16)));
typedef float  v8f   __attribute__((ext_vector_type(8)));
typedef float  v4f   __attribute__((ext_vector_type(4)));

union FragU { v16bf v; bf16_t e[16]; };

__device__ __forceinline__ void cvt4(bf16_t* d, v4f s) {
  // hardware f32->bf16 converts (v_cvt_pk_bf16_f32 pairs)
  d[0] = (bf16_t)s.x; d[1] = (bf16_t)s.y;
  d[2] = (bf16_t)s.z; d[3] = (bf16_t)s.w;
}

// A fragment (ISA 7.12.2, 16-bit A 16x32): this lane's 8 values at p[0..7]
// and 8 values at p[16..23] (p already includes the hi-lane +8 K offset).
__device__ __forceinline__ v16bf load_frag_a(const float* __restrict__ p) {
  FragU f;
  cvt4(f.e + 0,  *(const v4f*)(p + 0));
  cvt4(f.e + 4,  *(const v4f*)(p + 4));
  cvt4(f.e + 8,  *(const v4f*)(p + 16));
  cvt4(f.e + 12, *(const v4f*)(p + 20));
  return f.v;
}
// B fragment: 8 K values at p[0..7] and 8 at p[16..23] of one W row (B = W^T column).
__device__ __forceinline__ v16bf load_frag_b(const float* __restrict__ p) {
  FragU f;
  cvt4(f.e + 0,  *(const v4f*)(p + 0));
  cvt4(f.e + 4,  *(const v4f*)(p + 4));
  cvt4(f.e + 8,  *(const v4f*)(p + 16));
  cvt4(f.e + 12, *(const v4f*)(p + 20));
  return f.v;
}

// =====================================================================
// GEMM:  C[M,N] = act( A[M,K] @ W[N,K]^T + bias[N] )
// Wave tile 32x32 (2x2 WMMA tiles, fragments reused), block = 8 waves
// = 64(M) x 128(N).  Requires M%64==0, N%128==0, K%32==0 (all calls obey).
// act: 0 = none, 1 = relu, 2 = soft_sign
// =====================================================================
__global__ __launch_bounds__(256)
void k_gemm_bf16(const float* __restrict__ A, const float* __restrict__ W,
                 const float* __restrict__ bias, float* __restrict__ C,
                 int Ndim, int Kdim, int act)
{
  const int lane = threadIdx.x & 31;
  const int wave = threadIdx.x >> 5;
  const int wx = wave & 3, wy = wave >> 2;
  const int n0 = (blockIdx.x * 4 + wx) * 32;
  const int m0 = (blockIdx.y * 2 + wy) * 32;
  const int la = lane & 15;
  const int hi = lane >> 4;

  const float* a0 = A + (size_t)(m0 + la) * Kdim + (hi ? 8 : 0);
  const float* a1 = a0 + (size_t)16 * Kdim;
  const float* b0 = W + (size_t)(n0 + la) * Kdim + (hi ? 8 : 0);
  const float* b1 = b0 + (size_t)16 * Kdim;

  v8f acc00 = {0,0,0,0,0,0,0,0}, acc01 = {0,0,0,0,0,0,0,0};
  v8f acc10 = {0,0,0,0,0,0,0,0}, acc11 = {0,0,0,0,0,0,0,0};

#pragma unroll 2
  for (int k0 = 0; k0 < Kdim; k0 += 32) {
    __builtin_prefetch(a0 + k0 + 64, 0, 3);   // -> global_prefetch_b8
    __builtin_prefetch(b0 + k0 + 64, 0, 3);
    v16bf fa0 = load_frag_a(a0 + k0);
    v16bf fa1 = load_frag_a(a1 + k0);
    v16bf fb0 = load_frag_b(b0 + k0);
    v16bf fb1 = load_frag_b(b1 + k0);
    acc00 = __builtin_amdgcn_wmma_f32_16x16x32_bf16(false, fa0, false, fb0,
                                                    (short)0, acc00, false, false);
    acc01 = __builtin_amdgcn_wmma_f32_16x16x32_bf16(false, fa0, false, fb1,
                                                    (short)0, acc01, false, false);
    acc10 = __builtin_amdgcn_wmma_f32_16x16x32_bf16(false, fa1, false, fb0,
                                                    (short)0, acc10, false, false);
    acc11 = __builtin_amdgcn_wmma_f32_16x16x32_bf16(false, fa1, false, fb1,
                                                    (short)0, acc11, false, false);
  }

  const int col = n0 + la;
  const float bv0 = bias[col];
  const float bv1 = bias[col + 16];
#pragma unroll
  for (int r = 0; r < 8; ++r) {
    const int row = m0 + r + (hi ? 8 : 0);    // C layout: lanes 0-15 -> M=r, 16-31 -> M=r+8
    float v00 = acc00[r] + bv0, v01 = acc01[r] + bv1;
    float v10 = acc10[r] + bv0, v11 = acc11[r] + bv1;
    if (act == 1) {
      v00 = fmaxf(v00, 0.f); v01 = fmaxf(v01, 0.f);
      v10 = fmaxf(v10, 0.f); v11 = fmaxf(v11, 0.f);
    } else if (act == 2) {
      v00 /= 1.f + fabsf(v00); v01 /= 1.f + fabsf(v01);
      v10 /= 1.f + fabsf(v10); v11 /= 1.f + fabsf(v11);
    }
    C[(size_t)row * Ndim + col]             = v00;
    C[(size_t)row * Ndim + col + 16]        = v01;
    C[(size_t)(row + 16) * Ndim + col]      = v10;
    C[(size_t)(row + 16) * Ndim + col + 16] = v11;
  }
}

// =====================================================================
// Embedding gather + sinusoidal positional encoding
// =====================================================================
__global__ __launch_bounds__(256)
void k_embed(const int* __restrict__ src, const float* __restrict__ emb,
             float* __restrict__ X)
{
  const int t = blockIdx.x;
  const int l = t % SEQL;
  const int tok = src[t];
  const float ln10k = 9.210340371976184f;   // log(10000)
  for (int d = threadIdx.x; d < DMODEL; d += blockDim.x) {
    float ang = (float)l * expf((-2.0f * (float)d / (float)DMODEL) * ln10k);
    float pe = (d & 1) ? cosf(ang) : sinf(ang);
    X[(size_t)t * DMODEL + d] = emb[(size_t)tok * DMODEL + d] + pe;
  }
}

// =====================================================================
// Multi-head attention core: one block per (batch, head).  K/V staged
// into LDS with CDNA5 async loads (global_load_async_to_lds_b128,
// ASYNCcnt), then per-wave softmax(QK^T/8 + mask) @ V on VALU.
// =====================================================================
__global__ __launch_bounds__(256)
void k_attn(const float* __restrict__ qp, int qs,
            const float* __restrict__ kp, const float* __restrict__ vp, int kvs,
            float* __restrict__ out, int causal)
{
  const int b = blockIdx.x / NHEAD;
  const int h = blockIdx.x % NHEAD;
  __shared__ __align__(16) float sK[SEQL][DHEAD];
  __shared__ __align__(16) float sV[SEQL][DHEAD];
  __shared__ float sP[8][SEQL];
  const size_t base = (size_t)b * SEQL;

  // ---- async stage K and V tiles (60x64 f32 each) into LDS ----
  const unsigned ldsK = (unsigned)(size_t)&sK[0][0];   // LDS offset = addr[31:0]
  const unsigned ldsV = (unsigned)(size_t)&sV[0][0];
  for (int i = threadIdx.x; i < (SEQL * DHEAD) / 4; i += blockDim.x) {
    const int r  = i >> 4;             // 16 float4 per 64-wide row
    const int c4 = (i & 15) * 4;
    const float* gk = kp + (base + r) * (size_t)kvs + h * DHEAD + c4;
    const float* gv = vp + (base + r) * (size_t)kvs + h * DHEAD + c4;
    asm volatile("global_load_async_to_lds_b128 %0, %1, off"
                 :: "v"(ldsK + (unsigned)i * 16u), "v"(gk) : "memory");
    asm volatile("global_load_async_to_lds_b128 %0, %1, off"
                 :: "v"(ldsV + (unsigned)i * 16u), "v"(gv) : "memory");
  }
  asm volatile("s_wait_asynccnt 0x0" ::: "memory");
  __syncthreads();

  const int wave = threadIdx.x >> 5;
  const int lane = threadIdx.x & 31;
  for (int q = wave; q < SEQL; q += 8) {
    const float* qrow = qp + (base + q) * (size_t)qs + h * DHEAD;
    float s0 = 0.f, s1 = 0.f;
    for (int d = 0; d < DHEAD; ++d) {
      float qd = qrow[d];
      s0 += qd * sK[lane][d];
      if (lane + 32 < SEQL) s1 += qd * sK[lane + 32][d];
    }
    s0 *= 0.125f; s1 *= 0.125f;               // 1/sqrt(64)
    if (causal) {
      if (lane > q)      s0 = -1e30f;
      if (lane + 32 > q) s1 = -1e30f;
    }
    if (lane + 32 >= SEQL) s1 = -1e30f;

    float m = fmaxf(s0, s1);
    for (int off = 16; off > 0; off >>= 1) m = fmaxf(m, __shfl_xor(m, off, 32));
    float e0 = expf(s0 - m);
    float e1 = (lane + 32 < SEQL) ? expf(s1 - m) : 0.f;
    float sum = e0 + e1;
    for (int off = 16; off > 0; off >>= 1) sum += __shfl_xor(sum, off, 32);
    float inv = 1.f / sum;
    sP[wave][lane] = e0 * inv;
    if (lane + 32 < SEQL) sP[wave][lane + 32] = e1 * inv;

    for (int dd = lane; dd < DHEAD; dd += 32) {
      float o = 0.f;
      for (int j = 0; j < SEQL; ++j) o += sP[wave][j] * sV[j][dd];
      out[(base + q) * DMODEL + h * DHEAD + dd] = o;   // concat heads
    }
  }
}

// =====================================================================
// LayerNorm with fused residual:  out = LN(x + f) * g + b   (D = 512)
// =====================================================================
__device__ __forceinline__ float block_sum256(float v, float* red) {
  for (int off = 16; off > 0; off >>= 1) v += __shfl_xor(v, off, 32);
  const int wave = threadIdx.x >> 5, lane = threadIdx.x & 31;
  __syncthreads();
  if (lane == 0) red[wave] = v;
  __syncthreads();
  float s = 0.f;
  for (int i = 0; i < 8; ++i) s += red[i];
  return s;
}

__global__ __launch_bounds__(256)
void k_ln_res(const float* __restrict__ x, const float* __restrict__ f,
              const float* __restrict__ g, const float* __restrict__ bb,
              float* __restrict__ out)
{
  __shared__ float red[8];
  const size_t row = (size_t)blockIdx.x * DMODEL;
  const int t = threadIdx.x;
  float v0 = x[row + t]       + f[row + t];
  float v1 = x[row + t + 256] + f[row + t + 256];
  float mean = block_sum256(v0 + v1, red) * (1.f / DMODEL);
  float d0 = v0 - mean, d1 = v1 - mean;
  float var = block_sum256(d0 * d0 + d1 * d1, red) * (1.f / DMODEL);
  float rstd = rsqrtf(var + 1e-5f);
  out[row + t]       = d0 * rstd * g[t]       + bb[t];
  out[row + t + 256] = d1 * rstd * g[t + 256] + bb[t + 256];
}

// ---------------- small utility kernels ----------------
__global__ __launch_bounds__(256)
void k_gather_last(const float* __restrict__ X, float* __restrict__ Hb) {
  const int b = blockIdx.x;
  const size_t srow = ((size_t)b * SEQL + SEQL - 1) * DMODEL;
  for (int d = threadIdx.x; d < DMODEL; d += blockDim.x)
    Hb[(size_t)b * DMODEL + d] = X[srow + d];
}

__global__ __launch_bounds__(256)
void k_z(const float* __restrict__ mu, const float* __restrict__ lv,
         const float* __restrict__ eps, float* __restrict__ z) {
  const int i = blockIdx.x * blockDim.x + threadIdx.x;
  z[i] = mu[i] + eps[i] * expf(0.5f * lv[i]);
}

__global__ __launch_bounds__(256)
void k_bcast(const float* __restrict__ zd, float* __restrict__ mem) {
  const int t = blockIdx.x;
  const int b = t / SEQL;
  for (int d = threadIdx.x; d < DMODEL; d += blockDim.x)
    mem[(size_t)t * DMODEL + d] = zd[(size_t)b * DMODEL + d];
}

// =====================================================================
// Fused logits + argmax: argmax(softmax(l)) == argmax(l); ties -> lowest
// index (matches jnp.argmax).  Avoids a 154 MB logits buffer.
// =====================================================================
__global__ __launch_bounds__(256)
void k_argmax_logits(const float* __restrict__ Y, const float* __restrict__ Wf,
                     const float* __restrict__ bf, float* __restrict__ xhat)
{
  __shared__ __align__(16) float sy[DMODEL];
  __shared__ float bv[256];
  __shared__ int   bix[256];
  const int t = blockIdx.x;
  for (int d = threadIdx.x; d < DMODEL; d += blockDim.x)
    sy[d] = Y[(size_t)t * DMODEL + d];
  __syncthreads();
  float best = -3.4e38f; int bi = 0x7fffffff;
  for (int v = threadIdx.x; v < VOCAB; v += 256) {
    const float* w = Wf + (size_t)v * DMODEL;
    float s = bf[v];
    for (int d = 0; d < DMODEL; ++d) s += sy[d] * w[d];
    if (s > best) { best = s; bi = v; }   // v increasing -> first max kept
  }
  bv[threadIdx.x] = best; bix[threadIdx.x] = bi;
  __syncthreads();
  for (int off = 128; off > 0; off >>= 1) {
    if (threadIdx.x < off) {
      float ov = bv[threadIdx.x + off]; int oi = bix[threadIdx.x + off];
      if (ov > bv[threadIdx.x] ||
          (ov == bv[threadIdx.x] && oi < bix[threadIdx.x])) {
        bv[threadIdx.x] = ov; bix[threadIdx.x] = oi;
      }
    }
    __syncthreads();
  }
  if (threadIdx.x == 0) xhat[t] = (float)bix[0];
}

// =====================================================================
// Host: parameter unpacking (params dict flattened in insertion order)
// =====================================================================
struct MHAP { const float *wi, *bi, *wo, *bo; };
struct EncP { MHAP sa; const float *w1,*b1,*w2,*b2,*ln1g,*ln1b,*ln2g,*ln2b; };
struct DecP { MHAP sa, ca; const float *w1,*b1,*w2,*b2,
              *ln1g,*ln1b,*ln2g,*ln2b,*ln3g,*ln3b; };

extern "C" void kernel_launch(void* const* d_in, const int* in_sizes, int n_in,
                              void* d_out, int out_size, void* d_ws, size_t ws_size,
                              hipStream_t stream)
{
  (void)in_sizes; (void)n_in; (void)out_size; (void)ws_size;
  int idx = 0;
  auto F = [&]() { return (const float*)d_in[idx++]; };

  const int*   src = (const int*)d_in[idx++];
  const float* eps = F();
  const float* emb = F();
  EncP enc[NLAYER]; DecP dec[NLAYER];
  for (int i = 0; i < NLAYER; ++i) {
    EncP& p = enc[i];
    p.sa.wi=F(); p.sa.bi=F(); p.sa.wo=F(); p.sa.bo=F();
    p.w1=F(); p.b1=F(); p.w2=F(); p.b2=F();
    p.ln1g=F(); p.ln1b=F(); p.ln2g=F(); p.ln2b=F();
  }
  for (int i = 0; i < NLAYER; ++i) {
    DecP& p = dec[i];
    p.sa.wi=F(); p.sa.bi=F(); p.sa.wo=F(); p.sa.bo=F();
    p.ca.wi=F(); p.ca.bi=F(); p.ca.wo=F(); p.ca.bo=F();
    p.w1=F(); p.b1=F(); p.w2=F(); p.b2=F();
    p.ln1g=F(); p.ln1b=F(); p.ln2g=F(); p.ln2b=F(); p.ln3g=F(); p.ln3b=F();
  }
  const float* mu_w=F(); const float* mu_b=F();
  const float* lv_w=F(); const float* lv_b=F();
  const float* z_w =F(); const float* z_b =F();
  const float* fc_w=F(); const float* fc_b=F();

  // -------- workspace layout (floats) --------
  float* ws   = (float*)d_ws;
  float* X    = ws;                               // NTOK*D  activation (in place)
  float* AT   = X    + (size_t)NTOK * DMODEL;     // NTOK*D  attn concat out
  float* PR   = AT   + (size_t)NTOK * DMODEL;     // NTOK*D  proj / ffn out
  float* QC   = PR   + (size_t)NTOK * DMODEL;     // NTOK*D  cross-attn Q
  float* MEMB = QC   + (size_t)NTOK * DMODEL;     // NTOK*D  broadcast latent
  float* BIG  = MEMB + (size_t)NTOK * DMODEL;     // NTOK*FF QKV / FF hidden / KV
  float* Hb   = BIG  + (size_t)NTOK * FFDIM;      // B*D
  float* Zb   = Hb   + (size_t)BATCH * DMODEL;    // B*Z
  float* ZD   = Zb   + (size_t)BATCH * ZDIM;      // B*D

  float* xhat = (float*)d_out;                    // B*L
  float* mu_o = xhat + NTOK;                      // B*Z
  float* lv_o = mu_o + (size_t)BATCH * ZDIM;      // B*Z

  auto gemm = [&](const float* A, const float* W, const float* bias, float* Cm,
                  int Md, int Nd, int Kd, int act) {
    dim3 g(Nd / 128, Md / 64);
    k_gemm_bf16<<<g, dim3(256), 0, stream>>>(A, W, bias, Cm, Nd, Kd, act);
  };

  k_embed<<<NTOK, 256, 0, stream>>>(src, emb, X);

  // -------- encoder --------
  for (int i = 0; i < NLAYER; ++i) {
    const EncP& p = enc[i];
    gemm(X, p.sa.wi, p.sa.bi, BIG, NTOK, 3 * DMODEL, DMODEL, 0);        // QKV
    k_attn<<<BATCH * NHEAD, 256, 0, stream>>>(BIG, 3 * DMODEL,
        BIG + DMODEL, BIG + 2 * DMODEL, 3 * DMODEL, AT, 0);
    gemm(AT, p.sa.wo, p.sa.bo, PR, NTOK, DMODEL, DMODEL, 0);            // proj
    k_ln_res<<<NTOK, 256, 0, stream>>>(X, PR, p.ln1g, p.ln1b, X);
    gemm(X, p.w1, p.b1, BIG, NTOK, FFDIM, DMODEL, 1);                   // FF1+relu
    gemm(BIG, p.w2, p.b2, PR, NTOK, DMODEL, FFDIM, 0);                  // FF2
    k_ln_res<<<NTOK, 256, 0, stream>>>(X, PR, p.ln2g, p.ln2b, X);
  }

  // -------- latent head --------
  k_gather_last<<<BATCH, 256, 0, stream>>>(X, Hb);
  gemm(Hb, mu_w, mu_b, mu_o, BATCH, ZDIM, DMODEL, 2);                   // soft_sign
  gemm(Hb, lv_w, lv_b, lv_o, BATCH, ZDIM, DMODEL, 2);                   // soft_sign
  k_z<<<(BATCH * ZDIM) / 256, 256, 0, stream>>>(mu_o, lv_o, eps, Zb);
  gemm(Zb, z_w, z_b, ZD, BATCH, DMODEL, ZDIM, 2);                       // soft_sign
  k_bcast<<<NTOK, 256, 0, stream>>>(ZD, MEMB);

  // -------- decoder --------
  for (int i = 0; i < NLAYER; ++i) {
    const DecP& p = dec[i];
    gemm(X, p.sa.wi, p.sa.bi, BIG, NTOK, 3 * DMODEL, DMODEL, 0);        // QKV
    k_attn<<<BATCH * NHEAD, 256, 0, stream>>>(BIG, 3 * DMODEL,
        BIG + DMODEL, BIG + 2 * DMODEL, 3 * DMODEL, AT, 1);             // causal
    gemm(AT, p.sa.wo, p.sa.bo, PR, NTOK, DMODEL, DMODEL, 0);
    k_ln_res<<<NTOK, 256, 0, stream>>>(X, PR, p.ln1g, p.ln1b, X);
    gemm(X, p.ca.wi, p.ca.bi, QC, NTOK, DMODEL, DMODEL, 0);             // cross Q
    gemm(MEMB, p.ca.wi + (size_t)DMODEL * DMODEL, p.ca.bi + DMODEL,
         BIG, NTOK, 2 * DMODEL, DMODEL, 0);                             // cross KV
    k_attn<<<BATCH * NHEAD, 256, 0, stream>>>(QC, DMODEL,
        BIG, BIG + DMODEL, 2 * DMODEL, AT, 0);
    gemm(AT, p.ca.wo, p.ca.bo, PR, NTOK, DMODEL, DMODEL, 0);
    k_ln_res<<<NTOK, 256, 0, stream>>>(X, PR, p.ln2g, p.ln2b, X);
    gemm(X, p.w1, p.b1, BIG, NTOK, FFDIM, DMODEL, 1);
    gemm(BIG, p.w2, p.b2, PR, NTOK, DMODEL, FFDIM, 0);
    k_ln_res<<<NTOK, 256, 0, stream>>>(X, PR, p.ln3g, p.ln3b, X);
  }

  // -------- logits + argmax --------
  k_argmax_logits<<<NTOK, 256, 0, stream>>>(X, fc_w, fc_b, xhat);
}
